// LayerNormLSTMCell_47742856462402
// MI455X (gfx1250) — matrix operand order repaired and checked by reading, not compiled, not run on hardware
//
#include <hip/hip_runtime.h>
#include <hip/hip_bf16.h>

typedef __attribute__((ext_vector_type(16))) _Float16 v16h;
typedef __attribute__((ext_vector_type(8)))  float    v8f;
typedef __attribute__((ext_vector_type(4)))  unsigned int u32x4;
typedef __attribute__((ext_vector_type(4)))  unsigned int v4u;
typedef __attribute__((ext_vector_type(8)))  int          v8i;
typedef __attribute__((ext_vector_type(4)))  int          v4i;

#define BATCH 4096
#define HDIM  1024
#define KDIM  2048
#define NDIM  4096

// Use the Tensor Data Mover when the toolchain exposes it (clang-23 6-arg form).
#if defined(__gfx1250__) && __has_builtin(__builtin_amdgcn_tensor_load_to_lds) && \
    __has_builtin(__builtin_amdgcn_s_wait_tensorcnt)
#define USE_TDM 1
#else
#define USE_TDM 0
#endif

// ---------------- kernel 1: pack A = [h | x] as f16, row-major (B x K) ----------
__global__ __launch_bounds__(256) void pack_A(const float* __restrict__ x,
                                              const float* __restrict__ h,
                                              _Float16* __restrict__ A) {
    int t   = blockIdx.x * 256 + threadIdx.x;   // 0 .. B*K-1
    int row = t >> 11;                           // K = 2048
    int k   = t & 2047;
    float v = (k < 1024) ? h[row * 1024 + k] : x[row * 1024 + (k - 1024)];
    A[t] = (_Float16)v;
}

// ---------------- kernel 2: Wt[n][k] = ([W_h;W_x])[k][n] as f16 (N x K) ---------
__global__ __launch_bounds__(256) void pack_W(const float* __restrict__ Wh,
                                              const float* __restrict__ Wx,
                                              _Float16* __restrict__ Wt) {
    __shared__ _Float16 tile[32][33];
    int tn = blockIdx.x * 32;   // n tile
    int tk = blockIdx.y * 32;   // k tile
    int tx = threadIdx.x;       // 0..31
    int ty = threadIdx.y;       // 0..7
    for (int yy = 0; yy < 4; ++yy) {
        int k = tk + ty + yy * 8;
        int n = tn + tx;
        float v = (k < 1024) ? Wh[k * 4096 + n] : Wx[(k - 1024) * 4096 + n];
        tile[ty + yy * 8][tx] = (_Float16)v;
    }
    __syncthreads();
    for (int yy = 0; yy < 4; ++yy) {
        int n = tn + ty + yy * 8;
        int k = tk + tx;
        Wt[(long)n * KDIM + k] = tile[tx][ty + yy * 8];
    }
}

// ---------------- kernel 3: WMMA GEMM: C[B x N] = A[B x K] * W[K x N] -----------
union F16Frag { v16h v; u32x4 q[2]; };

#if USE_TDM
// Issue one TDM 2-D tile load: 128 rows x 32 halfs, row stride 2048 halfs.
// D# layout per CDNA5 ISA ch.8 (groups 0/1; groups 2/3 zero => 2-D tensor).
__device__ __forceinline__ void tdm_tile_load(unsigned lds_addr, const _Float16* gptr) {
    unsigned long long ga = (unsigned long long)gptr;
    v4u g0;
    g0[0] = 1u;                                          // count=1, user mode
    g0[1] = lds_addr;                                    // LDS byte address
    g0[2] = (unsigned)(ga & 0xffffffffull);              // global_addr[31:0]
    g0[3] = (unsigned)((ga >> 32) & 0x01ffffffull)       // global_addr[56:32]
            | (2u << 30);                                // type = 2 ("image")
    v8i g1;
    g1[0] = (1 << 16);            // workgroup_mask=0, data_size=1 (2 bytes)
    g1[1] = (32 << 16);           // abar=0, tensor_dim0[15:0]=32
    g1[2] = (128 << 16);          // tensor_dim0 hi=0, tensor_dim1[15:0]=128
    g1[3] = (32 << 16);           // tensor_dim1 hi=0, tile_dim0=32
    g1[4] = 128;                  // tile_dim1=128, tile_dim2=0
    g1[5] = 2048;                 // tensor_dim0_stride lo32 = 2048
    g1[6] = 0;                    // stride0 hi, stride1 lo
    g1[7] = 0;                    // stride1 hi
    v4i g2 = {0, 0, 0, 0};
    v4i g3 = {0, 0, 0, 0};
    v8i g4 = {0, 0, 0, 0, 0, 0, 0, 0};   // clang-23 6-arg form: extra group, zeroed
    __builtin_amdgcn_tensor_load_to_lds(g0, g1, g2, g3, g4, 0);
}
#endif

__global__ __launch_bounds__(256) void gemm_wmma(const _Float16* __restrict__ A,
                                                 const _Float16* __restrict__ Wt,
                                                 float* __restrict__ C) {
    __shared__ alignas(16) _Float16 As[2][128][32];
    __shared__ alignas(16) _Float16 Bs[2][128][32];   // N-major: Bs[b][n][k]

    const int tid   = threadIdx.x;
    const int lane  = tid & 31;
    const int wave  = tid >> 5;       // 0..7
    const int waveM = wave >> 2;      // 0..1  -> 64 rows each
    const int waveN = wave & 3;       // 0..3  -> 32 cols each
    const int tileM = blockIdx.y * 128;
    const int tileN = blockIdx.x * 128;

    const int hi   = lane >> 4;        // 0/1 half-wave
    const int lm   = lane & 15;
    const int kloA = hi * 8;           // A frag: K = klo..klo+7 and klo+16..klo+23
    const int kloB = hi * 8;           // B frag: K = klo..klo+7 and klo+16..klo+23

    v8f vzero = {};
    v8f acc[4][2];
    for (int i = 0; i < 4; ++i)
        for (int j = 0; j < 2; ++j) acc[i][j] = vzero;

#if USE_TDM
    const _Float16* gA = A  + (long)tileM * KDIM;   // +kt per step
    const _Float16* gB = Wt + (long)tileN * KDIM;
    const unsigned ldsA[2] = {(unsigned)(size_t)&As[0][0][0], (unsigned)(size_t)&As[1][0][0]};
    const unsigned ldsB[2] = {(unsigned)(size_t)&Bs[0][0][0], (unsigned)(size_t)&Bs[1][0][0]};

    if (tid < 32) {                    // wave 0 drives the TDM
        tdm_tile_load(ldsA[0], gA);
        tdm_tile_load(ldsB[0], gB);
    }
    int cur = 0;
    for (int kt = 0; kt < KDIM; kt += 32) {
        if (tid < 32) __builtin_amdgcn_s_wait_tensorcnt(0);
        __syncthreads();               // current buffers are resident
        if (kt + 32 < KDIM && tid < 32) {
            tdm_tile_load(ldsA[cur ^ 1], gA + kt + 32);   // prefetch next tile
            tdm_tile_load(ldsB[cur ^ 1], gB + kt + 32);   // overlaps WMMA below
        }
        F16Frag af[4], bf[2];
        for (int i = 0; i < 4; ++i) {
            int r = waveM * 64 + i * 16 + lm;
            af[i].q[0] = *(const u32x4*)&As[cur][r][kloA];
            af[i].q[1] = *(const u32x4*)&As[cur][r][kloA + 16];
        }
        for (int j = 0; j < 2; ++j) {
            int n = waveN * 32 + j * 16 + lm;
            bf[j].q[0] = *(const u32x4*)&Bs[cur][n][kloB];
            bf[j].q[1] = *(const u32x4*)&Bs[cur][n][kloB + 16];
        }
        for (int i = 0; i < 4; ++i)
            for (int j = 0; j < 2; ++j)
                acc[i][j] = __builtin_amdgcn_wmma_f32_16x16x32_f16(
                    false, af[i].v, false, bf[j].v, (short)0, acc[i][j], false, false);
        cur ^= 1;
    }
#else
    const int lrow  = tid >> 1;        // 0..127 (staging row)
    const int lhalf = (tid & 1) * 16;  // 0 or 16 halves
    for (int kt = 0; kt < KDIM; kt += 32) {
        const u32x4* ga = (const u32x4*)(A  + (long)(tileM + lrow) * KDIM + kt + lhalf);
        const u32x4* gb = (const u32x4*)(Wt + (long)(tileN + lrow) * KDIM + kt + lhalf);
        u32x4 va0 = ga[0], va1 = ga[1];
        u32x4 vb0 = gb[0], vb1 = gb[1];
        __syncthreads();
        *(u32x4*)&As[0][lrow][lhalf]     = va0;
        *(u32x4*)&As[0][lrow][lhalf + 8] = va1;
        *(u32x4*)&Bs[0][lrow][lhalf]     = vb0;
        *(u32x4*)&Bs[0][lrow][lhalf + 8] = vb1;
        __syncthreads();
        if (kt + 32 < KDIM) {   // near-scope prefetch of next K tile
            __builtin_prefetch(A  + (long)(tileM + lrow) * KDIM + kt + 32 + lhalf, 0, 3);
            __builtin_prefetch(Wt + (long)(tileN + lrow) * KDIM + kt + 32 + lhalf, 0, 3);
        }
        F16Frag af[4], bf[2];
        for (int i = 0; i < 4; ++i) {
            int r = waveM * 64 + i * 16 + lm;
            af[i].q[0] = *(const u32x4*)&As[0][r][kloA];
            af[i].q[1] = *(const u32x4*)&As[0][r][kloA + 16];
        }
        for (int j = 0; j < 2; ++j) {
            int n = waveN * 32 + j * 16 + lm;
            bf[j].q[0] = *(const u32x4*)&Bs[0][n][kloB];
            bf[j].q[1] = *(const u32x4*)&Bs[0][n][kloB + 16];
        }
        for (int i = 0; i < 4; ++i)
            for (int j = 0; j < 2; ++j)
                acc[i][j] = __builtin_amdgcn_wmma_f32_16x16x32_f16(
                    false, af[i].v, false, bf[j].v, (short)0, acc[i][j], false, false);
    }
#endif

    // C/D layout: VGPR r -> M = r + 8*hi, N = lane&15
    for (int i = 0; i < 4; ++i)
        for (int j = 0; j < 2; ++j) {
            int col   = tileN + waveN * 32 + j * 16 + lm;
            int rbase = tileM + waveM * 64 + i * 16 + hi * 8;
            for (int r = 0; r < 8; ++r)
                C[(long)(rbase + r) * NDIM + col] = acc[i][j][r];
        }
}

// ---------------- kernel 4: fused bias + per-gate LN + gates + c-LN ------------
__device__ __forceinline__ float waveRed(float v) {
    for (int o = 16; o; o >>= 1) v += __shfl_xor(v, o, 32);
    return v;
}
__device__ __forceinline__ float sigm(float x) { return 1.0f / (1.0f + expf(-x)); }

__global__ __launch_bounds__(256) void ln_gates(const float* __restrict__ ifgo,
                                                const float* __restrict__ cin,
                                                const float* __restrict__ b_h,
                                                const float* __restrict__ ln_g,
                                                const float* __restrict__ ln_b,
                                                const float* __restrict__ lnc_g,
                                                const float* __restrict__ lnc_b,
                                                float* __restrict__ out) {
    const int row  = blockIdx.x;
    const int t    = threadIdx.x;     // 256 threads, 4 cols each
    const int lane = t & 31, wave = t >> 5;
    __shared__ float red[8][8];

    const float4* prow = (const float4*)(ifgo + (long)row * NDIM);
    const float4* b4   = (const float4*)b_h;

    float v[4][4];
    for (int g = 0; g < 4; ++g) {
        float4 a  = prow[g * 256 + t];
        float4 bb = b4[g * 256 + t];
        v[g][0] = a.x + bb.x; v[g][1] = a.y + bb.y;
        v[g][2] = a.z + bb.z; v[g][3] = a.w + bb.w;
        float s  = v[g][0] + v[g][1] + v[g][2] + v[g][3];
        float ss = v[g][0]*v[g][0] + v[g][1]*v[g][1] + v[g][2]*v[g][2] + v[g][3]*v[g][3];
        s = waveRed(s); ss = waveRed(ss);
        if (lane == 0) { red[wave][g] = s; red[wave][4 + g] = ss; }
    }
    __syncthreads();
    float mu[4], rs[4];
    for (int g = 0; g < 4; ++g) {
        float s = 0.f, ss = 0.f;
        for (int w = 0; w < 8; ++w) { s += red[w][g]; ss += red[w][4 + g]; }
        float m  = s * (1.0f / HDIM);
        float vr = ss * (1.0f / HDIM) - m * m;
        mu[g] = m; rs[g] = rsqrtf(vr + 1e-5f);
    }

    const float4* g4  = (const float4*)ln_g;
    const float4* be4 = (const float4*)ln_b;
    float ln[4][4];
    for (int g = 0; g < 4; ++g) {
        float4 gam = g4[g * 256 + t], bet = be4[g * 256 + t];
        ln[g][0] = (v[g][0] - mu[g]) * rs[g] * gam.x + bet.x;
        ln[g][1] = (v[g][1] - mu[g]) * rs[g] * gam.y + bet.y;
        ln[g][2] = (v[g][2] - mu[g]) * rs[g] * gam.z + bet.z;
        ln[g][3] = (v[g][3] - mu[g]) * rs[g] * gam.w + bet.w;
    }

    float4 cv = ((const float4*)(cin + (long)row * HDIM))[t];
    float cvv[4] = {cv.x, cv.y, cv.z, cv.w};
    float cn[4];
    for (int e = 0; e < 4; ++e)   // i=gate0, f=gate1, g=gate2, o=gate3
        cn[e] = sigm(ln[1][e]) * cvv[e] + sigm(ln[0][e]) * tanhf(ln[2][e]);
    ((float4*)(out + (long)BATCH * HDIM + (long)row * HDIM))[t] =
        make_float4(cn[0], cn[1], cn[2], cn[3]);

    float s  = cn[0] + cn[1] + cn[2] + cn[3];
    float ss = cn[0]*cn[0] + cn[1]*cn[1] + cn[2]*cn[2] + cn[3]*cn[3];
    s = waveRed(s); ss = waveRed(ss);
    __syncthreads();                 // protect red[] reuse
    if (lane == 0) { red[wave][0] = s; red[wave][1] = ss; }
    __syncthreads();
    float st = 0.f, sst = 0.f;
    for (int w = 0; w < 8; ++w) { st += red[w][0]; sst += red[w][1]; }
    float mc = st * (1.0f / HDIM);
    float rc = rsqrtf(sst * (1.0f / HDIM) - mc * mc + 1e-5f);

    float4 gc = ((const float4*)lnc_g)[t], bc = ((const float4*)lnc_b)[t];
    float gcv[4] = {gc.x, gc.y, gc.z, gc.w}, bcv[4] = {bc.x, bc.y, bc.z, bc.w};
    float hn[4];
    for (int e = 0; e < 4; ++e) {
        float lnc = (cn[e] - mc) * rc * gcv[e] + bcv[e];
        hn[e] = sigm(ln[3][e]) * tanhf(lnc);
    }
    ((float4*)(out + (long)row * HDIM))[t] = make_float4(hn[0], hn[1], hn[2], hn[3]);
}

// --------------------------------- launcher ------------------------------------
extern "C" void kernel_launch(void* const* d_in, const int* in_sizes, int n_in,
                              void* d_out, int out_size, void* d_ws, size_t ws_size,
                              hipStream_t stream) {
    const float* x     = (const float*)d_in[0];
    const float* h     = (const float*)d_in[1];
    const float* c     = (const float*)d_in[2];
    const float* W_h   = (const float*)d_in[3];
    const float* b_h   = (const float*)d_in[4];
    const float* W_x   = (const float*)d_in[5];
    const float* ln_g  = (const float*)d_in[6];
    const float* ln_b  = (const float*)d_in[7];
    const float* lnc_g = (const float*)d_in[8];
    const float* lnc_b = (const float*)d_in[9];
    float* out = (float*)d_out;

    _Float16* Apack = (_Float16*)d_ws;                       // 4096x2048 f16
    _Float16* Wt    = Apack + (size_t)BATCH * KDIM;          // 4096x2048 f16
    float*    ifgo  = (float*)(Wt + (size_t)NDIM * KDIM);    // 4096x4096 f32

    pack_A<<<(BATCH * KDIM) / 256, 256, 0, stream>>>(x, h, Apack);
    pack_W<<<dim3(NDIM / 32, KDIM / 32), dim3(32, 8), 0, stream>>>(W_h, W_x, Wt);
    gemm_wmma<<<dim3(NDIM / 128, BATCH / 128), 256, 0, stream>>>(Apack, Wt, ifgo);
    ln_gates<<<BATCH, 256, 0, stream>>>(ifgo, c, b_h, ln_g, ln_b, lnc_g, lnc_b, out);
}
